// MultiHeadAttention_51754355917424
// MI455X (gfx1250) — matrix operand order repaired and checked by reading, not compiled, not run on hardware
//
#include <hip/hip_runtime.h>
#include <hip/hip_bf16.h>

// Problem constants (from reference): B=2, S=2048, D=1024, H=16, DK=64
#define BB 2
#define SS 2048
#define DD 1024
#define HH 16
#define DKK 64
#define MTOT (BB * SS)          // 4096 rows in every projection GEMM

typedef __attribute__((ext_vector_type(16))) __bf16 v16bf;
typedef __attribute__((ext_vector_type(8)))  float  v8f;
typedef __attribute__((__vector_size__(4 * sizeof(int)))) int v4i;

union FragAB {
    v16bf v;
    uint4 u[2];
};

#define AS1 __attribute__((address_space(1)))
#define AS3 __attribute__((address_space(3)))

#if __has_builtin(__builtin_amdgcn_global_load_async_to_lds_b128) && \
    __has_builtin(__builtin_amdgcn_s_wait_asynccnt)
#define ASYNC_OK 1
#else
#define ASYNC_OK 0
#endif

// Copy 16 bytes global -> LDS. Async (ASYNCcnt-tracked) if available.
__device__ __forceinline__ void cp16(const unsigned short* g, unsigned short* l) {
#if ASYNC_OK
    __builtin_amdgcn_global_load_async_to_lds_b128((AS1 v4i*)g, (AS3 v4i*)l, 0, 0);
#else
    *(uint4*)l = *(const uint4*)g;
#endif
}
__device__ __forceinline__ void cp_wait() {
#if ASYNC_OK
    __builtin_amdgcn_s_wait_asynccnt(0);
#endif
}

__device__ __forceinline__ unsigned short f2bf(float f) {
    unsigned int u = __builtin_bit_cast(unsigned int, f);
    u = (u + 0x7FFFu + ((u >> 16) & 1u)) >> 16;   // round-to-nearest-even
    return (unsigned short)u;
}

// ---------------------------------------------------------------------------
// f32 -> bf16 conversion
// ---------------------------------------------------------------------------
__global__ void cvt_f32_bf16(const float* __restrict__ in,
                             unsigned short* __restrict__ out, int n) {
    int i = blockIdx.x * blockDim.x + threadIdx.x;
    if (i < n) out[i] = f2bf(in[i]);
}

// ---------------------------------------------------------------------------
// Projection GEMM:  Y[4096,1024] = X @ W^T + bias   (WMMA bf16, f32 acc)
// Block = 8 waves = 128x64 output tile. Per K-step (32), A(128x32) and
// B(64x32) tiles are async-staged into LDS, double-buffered; B tile shared
// by all 8 waves.
// mode 0: Q -> [B,H,S,DK] bf16, scaled 1/sqrt(DK)   mode 1: K -> [B,H,S,DK]
// mode 2: V -> [B,H,DK,S] bf16 (transposed)         mode 3: O -> [B,S,D] f32
// ---------------------------------------------------------------------------
__global__ __launch_bounds__(256)
void proj_gemm(const unsigned short* __restrict__ X,
               const unsigned short* __restrict__ W,
               const float* __restrict__ bias,
               void* __restrict__ out, int mode) {
    __shared__ unsigned short At[2][128 * 32];   // 2 x 8KB
    __shared__ unsigned short Bt[2][64 * 32];    // 2 x 4KB

    const int tid  = threadIdx.x;
    const int lane = tid & 31;
    const int lid  = lane & 15;
    const int half = lane >> 4;
    const int wid  = tid >> 5;
    const int mbase = (blockIdx.x & 31) * 128;   // 4096/128 = 32 row-blocks
    const int ngrp  = blockIdx.x >> 5;           // 0..15 (64-col group)

    auto stage = [&](int k0, int bsel) {
        // A tile: 512 16B chunks, 2 per thread. chunk c: row c>>2, k (c&3)*8
        const unsigned short* ga = X + (size_t)(mbase + (tid >> 2)) * DD + k0 + (tid & 3) * 8;
        cp16(ga, &At[bsel][tid * 8]);
        const int c2 = tid + 256;
        const unsigned short* ga2 = X + (size_t)(mbase + (c2 >> 2)) * DD + k0 + (c2 & 3) * 8;
        cp16(ga2, &At[bsel][c2 * 8]);
        // B tile: 256 chunks, 1 per thread. chunk: row(gn) tid>>2, k (tid&3)*8
        const unsigned short* gb = W + (size_t)(ngrp * 64 + (tid >> 2)) * DD + k0 + (tid & 3) * 8;
        cp16(gb, &Bt[bsel][tid * 8]);
    };

    stage(0, 0);
    v8f acc[4] = {};
    int buf = 0;
    for (int k0 = 0; k0 < DD; k0 += 32) {
        cp_wait();
        __syncthreads();                       // staged tile ready; prev reads done
        if (k0 + 32 < DD) stage(k0 + 32, buf ^ 1);

        FragAB af;                             // A-frag: chunks at half*8 and 16+half*8
        const unsigned short* pa = &At[buf][(wid * 16 + lid) * 32 + half * 8];
        af.u[0] = *(const uint4*)(pa);
        af.u[1] = *(const uint4*)(pa + 16);
#pragma unroll
        for (int nt = 0; nt < 4; ++nt) {       // B-frag: chunks at half*8 and 16+half*8
            FragAB bfr;
            const unsigned short* pb = &Bt[buf][(nt * 16 + lid) * 32 + half * 8];
            bfr.u[0] = *(const uint4*)(pb);
            bfr.u[1] = *(const uint4*)(pb + 16);
            acc[nt] = __builtin_amdgcn_wmma_f32_16x16x32_bf16(
                false, af.v, false, bfr.v, (short)0, acc[nt], false, false);
        }
        buf ^= 1;
    }

#pragma unroll
    for (int nt = 0; nt < 4; ++nt) {
        const int gn = ngrp * 64 + nt * 16 + lid;
        const float bs = bias[gn];
#pragma unroll
        for (int r = 0; r < 8; ++r) {
            const int gm = mbase + wid * 16 + r + 8 * half;   // token index b*S+s
            float v = acc[nt][r] + bs;
            if (mode == 0) {                            // Q, pre-scaled
                v *= 0.125f;                            // 1/sqrt(64)
                const int h = gn >> 6, dk = gn & 63;
                const int b = gm >> 11, s = gm & (SS - 1);
                ((unsigned short*)out)[(((size_t)(b * HH + h)) * SS + s) * DKK + dk] = f2bf(v);
            } else if (mode == 1) {                     // K
                const int h = gn >> 6, dk = gn & 63;
                const int b = gm >> 11, s = gm & (SS - 1);
                ((unsigned short*)out)[(((size_t)(b * HH + h)) * SS + s) * DKK + dk] = f2bf(v);
            } else if (mode == 2) {                     // V transposed
                const int h = gn >> 6, dk = gn & 63;
                const int b = gm >> 11, s = gm & (SS - 1);
                ((unsigned short*)out)[(((size_t)(b * HH + h)) * DKK + dk) * SS + s] = f2bf(v);
            } else {                                    // final O, f32
                ((float*)out)[(size_t)gm * DD + gn] = v;
            }
        }
    }
}

// ---------------------------------------------------------------------------
// Flash attention. Block = 8 waves sharing one (b,h); wave w owns query tile
// qt = (blockIdx&15)*8 + w (16 rows x DK=64). Per 32-key step, K(32x64) and
// VT(64x32) tiles are async-staged to LDS, double-buffered, shared by all
// 8 waves. Online softmax; P tile converted C-layout -> A-layout via LDS.
// ---------------------------------------------------------------------------
__global__ __launch_bounds__(256)
void attn_kernel(const unsigned short* __restrict__ Qh,
                 const unsigned short* __restrict__ Kh,
                 const unsigned short* __restrict__ VT,
                 unsigned short* __restrict__ AX) {
    __shared__ unsigned short Kt[2][32 * 64];     // 2 x 4KB
    __shared__ unsigned short Vt[2][64 * 32];     // 2 x 4KB
    __shared__ unsigned short plds[8][16 * 32];   // 8KB

    const int tid  = threadIdx.x;
    const int lane = tid & 31;
    const int lid  = lane & 15;
    const int half = lane >> 4;
    const int wid  = tid >> 5;
    const int bh   = blockIdx.x >> 4;             // all 8 waves share (b,h)
    const int qt   = (blockIdx.x & 15) * 8 + wid; // 0..127
    const int b    = bh >> 4, h = bh & 15;

    const unsigned short* Qb = Qh + ((size_t)bh * SS + qt * 16) * DKK;
    const unsigned short* Kb = Kh + (size_t)bh * SS * DKK;
    const unsigned short* Vb = VT + (size_t)bh * DKK * SS;

    auto stage = [&](int j, int bsel) {
        // K block: 32 keys x 64 dk. chunk tid: key tid>>3, dk (tid&7)*8
        const unsigned short* gk = Kb + (size_t)(j + (tid >> 3)) * DKK + (tid & 7) * 8;
        cp16(gk, &Kt[bsel][tid * 8]);
        // VT block: 64 dk x 32 keys. chunk tid: dk tid>>2, key j+(tid&3)*8
        const unsigned short* gv = Vb + (size_t)(tid >> 2) * SS + j + (tid & 3) * 8;
        cp16(gv, &Vt[bsel][tid * 8]);
    };

    // Q fragments (global, loaded once): two 16x32 A-fragments
    FragAB qf[2];
#pragma unroll
    for (int c = 0; c < 2; ++c) {
        const unsigned short* p = Qb + lid * DKK + c * 32 + half * 8;
        qf[c].u[0] = *(const uint4*)(p);
        qf[c].u[1] = *(const uint4*)(p + 16);
    }

    v8f acc[4] = {};
    float m[8], l[8];
#pragma unroll
    for (int r = 0; r < 8; ++r) { m[r] = -1e30f; l[r] = 0.f; }

    stage(0, 0);
    int buf = 0;
    for (int j = 0; j < SS; j += 32) {
        cp_wait();
        __syncthreads();                 // tiles ready; prev-iter LDS reads done
        if (j + 32 < SS) stage(j + 32, buf ^ 1);

        // ---- scores: two 16x16 tiles ----
        v8f sc[2];
#pragma unroll
        for (int nt = 0; nt < 2; ++nt) {
            v8f s = {};
#pragma unroll
            for (int c = 0; c < 2; ++c) {
                FragAB kf;    // B-frag: dk chunks at c*32 + half*8 and +16
                const unsigned short* p = &Kt[buf][(nt * 16 + lid) * 64 + c * 32 + half * 8];
                kf.u[0] = *(const uint4*)(p);
                kf.u[1] = *(const uint4*)(p + 16);
                s = __builtin_amdgcn_wmma_f32_16x16x32_bf16(
                    false, qf[c].v, false, kf.v, (short)0, s, false, false);
            }
            sc[nt] = s;
        }

        // ---- online softmax ----
        float p0[8], p1[8];
#pragma unroll
        for (int r = 0; r < 8; ++r) {
            float mx = fmaxf(sc[0][r], sc[1][r]);
            mx = fmaxf(mx, __shfl_xor(mx, 1, 16));
            mx = fmaxf(mx, __shfl_xor(mx, 2, 16));
            mx = fmaxf(mx, __shfl_xor(mx, 4, 16));
            mx = fmaxf(mx, __shfl_xor(mx, 8, 16));
            const float mn = fmaxf(m[r], mx);
            const float alpha = __expf(m[r] - mn);
            m[r] = mn;
            const float e0 = __expf(sc[0][r] - mn);
            const float e1 = __expf(sc[1][r] - mn);
            float rs = e0 + e1;
            rs += __shfl_xor(rs, 1, 16);
            rs += __shfl_xor(rs, 2, 16);
            rs += __shfl_xor(rs, 4, 16);
            rs += __shfl_xor(rs, 8, 16);
            l[r] = l[r] * alpha + rs;
#pragma unroll
            for (int t = 0; t < 4; ++t) acc[t][r] *= alpha;
            p0[r] = e0; p1[r] = e1;
        }

        // ---- C-layout -> A-fragment layout via LDS ----
        unsigned short* pl = plds[wid];
#pragma unroll
        for (int r = 0; r < 8; ++r) {
            pl[(r + 8 * half) * 32 + lid]      = f2bf(p0[r]);
            pl[(r + 8 * half) * 32 + 16 + lid] = f2bf(p1[r]);
        }
        __syncthreads();
        FragAB pf;
        {
            const unsigned short* p = pl + lid * 32 + half * 8;
            pf.u[0] = *(const uint4*)(p);
            pf.u[1] = *(const uint4*)(p + 16);
        }

        // ---- P @ V ----
#pragma unroll
        for (int t = 0; t < 4; ++t) {
            FragAB vf;    // B-frag: key chunks at half*8 and 16+half*8
            const unsigned short* p = &Vt[buf][(t * 16 + lid) * 32 + half * 8];
            vf.u[0] = *(const uint4*)(p);
            vf.u[1] = *(const uint4*)(p + 16);
            acc[t] = __builtin_amdgcn_wmma_f32_16x16x32_bf16(
                false, pf.v, false, vf.v, (short)0, acc[t], false, false);
        }
        buf ^= 1;
    }

    // ---- normalize + write back to [B,S,D] bf16 ----
#pragma unroll
    for (int t = 0; t < 4; ++t) {
#pragma unroll
        for (int r = 0; r < 8; ++r) {
            const int s = qt * 16 + r + 8 * half;
            const float o = acc[t][r] / l[r];
            AX[((size_t)b * SS + s) * DD + h * DKK + t * 16 + lid] = f2bf(o);
        }
    }
}

// ---------------------------------------------------------------------------
// Host-side orchestration
// ---------------------------------------------------------------------------
extern "C" void kernel_launch(void* const* d_in, const int* in_sizes, int n_in,
                              void* d_out, int out_size, void* d_ws, size_t ws_size,
                              hipStream_t stream) {
    const float* q  = (const float*)d_in[0];
    const float* k  = (const float*)d_in[1];
    const float* v  = (const float*)d_in[2];
    const float* Wq = (const float*)d_in[3];
    const float* bq = (const float*)d_in[4];
    const float* Wk = (const float*)d_in[5];
    const float* bk = (const float*)d_in[6];
    const float* Wv = (const float*)d_in[7];
    const float* bv = (const float*)d_in[8];
    const float* Wo = (const float*)d_in[9];
    const float* bo = (const float*)d_in[10];

    char* ws = (char*)d_ws;
    size_t off = 0;
    auto alloc = [&](size_t bytes) -> void* {
        void* p = ws + off;
        off += (bytes + 255) & ~(size_t)255;
        return p;
    };

    const size_t NX = (size_t)MTOT * DD;      // 4M elements
    const size_t NW = (size_t)DD * DD;        // 1M elements

    unsigned short* Xq  = (unsigned short*)alloc(NX * 2);
    unsigned short* Xk  = (unsigned short*)alloc(NX * 2);
    unsigned short* Xv  = (unsigned short*)alloc(NX * 2);
    unsigned short* Wqb = (unsigned short*)alloc(NW * 2);
    unsigned short* Wkb = (unsigned short*)alloc(NW * 2);
    unsigned short* Wvb = (unsigned short*)alloc(NW * 2);
    unsigned short* Wob = (unsigned short*)alloc(NW * 2);
    unsigned short* Qh  = (unsigned short*)alloc(NX * 2);  // [B,H,S,DK]
    unsigned short* Kh  = (unsigned short*)alloc(NX * 2);  // [B,H,S,DK]
    unsigned short* VT  = (unsigned short*)alloc(NX * 2);  // [B,H,DK,S]
    unsigned short* AX  = (unsigned short*)alloc(NX * 2);  // [B,S,D]

    // 1) convert inputs and weights to bf16
    cvt_f32_bf16<<<(int)((NX + 255) / 256), 256, 0, stream>>>(q, Xq, (int)NX);
    cvt_f32_bf16<<<(int)((NX + 255) / 256), 256, 0, stream>>>(k, Xk, (int)NX);
    cvt_f32_bf16<<<(int)((NX + 255) / 256), 256, 0, stream>>>(v, Xv, (int)NX);
    cvt_f32_bf16<<<(int)((NW + 255) / 256), 256, 0, stream>>>(Wq, Wqb, (int)NW);
    cvt_f32_bf16<<<(int)((NW + 255) / 256), 256, 0, stream>>>(Wk, Wkb, (int)NW);
    cvt_f32_bf16<<<(int)((NW + 255) / 256), 256, 0, stream>>>(Wv, Wvb, (int)NW);
    cvt_f32_bf16<<<(int)((NW + 255) / 256), 256, 0, stream>>>(Wo, Wob, (int)NW);

    // 2) QKV projections: 32 row-blocks x 16 col-groups = 512 blocks
    proj_gemm<<<512, 256, 0, stream>>>(Xq, Wqb, bq, Qh, 0);
    proj_gemm<<<512, 256, 0, stream>>>(Xk, Wkb, bk, Kh, 1);
    proj_gemm<<<512, 256, 0, stream>>>(Xv, Wvb, bv, VT, 2);

    // 3) flash attention: 32 (b,h) x 16 blocks of 8 query tiles
    attn_kernel<<<512, 256, 0, stream>>>(Qh, Kh, VT, AX);

    // 4) output projection -> f32 d_out
    proj_gemm<<<512, 256, 0, stream>>>(AX, Wob, bo, d_out, 3);
}
